// GraphAttentionBlock_7687991460578
// MI455X (gfx1250) — compile-verified, matrix operand order checked by reading, not run on hardware
//
#include <hip/hip_runtime.h>
#include <hip/hip_bf16.h>
#include <math.h>

// Problem constants (match reference)
#define BB   4
#define NNN  2048
#define DDD  256
#define HHH  8
#define HDD  32
#define DFF  1024
#define EPSLN 1e-5f

typedef __bf16 bf16_t;
typedef __attribute__((ext_vector_type(16))) __bf16 v16bf;
typedef __attribute__((ext_vector_type(8)))  float  v8f;

union Frag16 { v16bf v; __bf16 h[16]; unsigned int u[8]; };

__device__ inline unsigned short f2bf_bits(float x) {
  union { __bf16 b; unsigned short s; } u; u.b = (__bf16)x; return u.s;
}

// ---- DPP16 butterfly reductions over the 16-lane halves of a wave32 -------
// xor1 = quad_perm(1,0,3,2)=0xB1, xor2 = quad_perm(2,3,0,1)=0x4E,
// then row_half_mirror (i^7) and row_mirror (i^15) complete the tree once the
// lower groups are uniform.  DPP rows == the 16-lane halves that hold one row
// of the WMMA C layout, so no cross-half leakage.
template <int CTRL>
__device__ __forceinline__ float dpp_xor_f(float v) {
  int s = __builtin_bit_cast(int, v);
  int r = __builtin_amdgcn_update_dpp(s, s, CTRL, 0xF, 0xF, true);
  return __builtin_bit_cast(float, r);
}
__device__ __forceinline__ float row_max16(float x) {
  x = fmaxf(x, dpp_xor_f<0xB1>(x));
  x = fmaxf(x, dpp_xor_f<0x4E>(x));
  x = fmaxf(x, dpp_xor_f<0x141>(x));
  x = fmaxf(x, dpp_xor_f<0x140>(x));
  return x;
}
__device__ __forceinline__ float row_sum16(float x) {
  x += dpp_xor_f<0xB1>(x);
  x += dpp_xor_f<0x4E>(x);
  x += dpp_xor_f<0x141>(x);
  x += dpp_xor_f<0x140>(x);
  return x;
}

// Load a 16x32 bf16 fragment (A-layout; identical striping works for B when
// fed the transposed operand).  Per CDNA5 ISA 7.12.2:
//   lanes 0-15 : row = lane,     K = {0..7, 16..23}
//   lanes 16-31: row = lane-16,  K = {8..15, 24..31}
__device__ inline v16bf load_frag_g(const bf16_t* __restrict__ p, int ld) {
  int lane = threadIdx.x & 31;
  int row  = lane & 15;
  int koff = (lane >> 4) << 3;                 // 0 or 8
  const bf16_t* base = p + (size_t)row * ld + koff;
  Frag16 f;
#pragma unroll
  for (int i = 0; i < 8; ++i) {
    int k0 = ((i >> 2) << 4) + ((i & 3) << 1); // 0,2,4,6,16,18,20,22
    f.u[i] = *(const unsigned int*)(base + k0);
  }
  return f.v;
}

// Same fragment layout but sourced from a row-major 16x32 bf16 LDS tile.
__device__ inline v16bf load_frag_lds(const unsigned short* p) {
  int lane = threadIdx.x & 31;
  int row  = lane & 15;
  int koff = (lane >> 4) << 3;
  Frag16 f;
#pragma unroll
  for (int i = 0; i < 8; ++i) {
    int k0 = ((i >> 2) << 4) + ((i & 3) << 1);
    f.u[i] = *(const unsigned int*)(p + row * 32 + koff + k0);
  }
  return f.v;
}

__device__ inline v8f wmma_bf16(v16bf a, v16bf b, v8f c) {
  return __builtin_amdgcn_wmma_f32_16x16x32_bf16(false, a, false, b,
                                                 (short)0, c, false, false);
}

// ---------------- weight convert + transpose: f32 KxN -> bf16 NxK ----------
__global__ void wtrans_kernel(const float* __restrict__ w, bf16_t* __restrict__ wt,
                              int K, int Nn) {
  int idx = blockIdx.x * 256 + threadIdx.x;
  if (idx >= K * Nn) return;
  int n  = idx / K;
  int kk = idx - n * K;
  wt[idx] = (bf16_t)w[(size_t)kk * Nn + n];
}

// ---------------- adjacency -> bitmask: 64MB int32 -> 2MB bits -------------
// bits[b][n][w] bit j = (adj[b][n][w*32+j] != 0).  One u32 word == one 32-key
// attention tile, so the mask for a whole tile row is a single broadcast load.
__global__ void adjpack_kernel(const int* __restrict__ adj,
                               unsigned int* __restrict__ bits) {
  int w = blockIdx.x * 256 + threadIdx.x;      // word index over B*N*(N/32)
  const int* p = adj + (size_t)w * 32;
  unsigned int m = 0;
#pragma unroll
  for (int j = 0; j < 32; ++j) m |= (p[j] != 0 ? 1u : 0u) << j;
  bits[w] = m;
}

// ---------------- V repack: (B,N,D) bf16 -> per-head (B,H,HD,N) ------------
__global__ void vtrans_kernel(const bf16_t* __restrict__ v, bf16_t* __restrict__ vt) {
  int idx = blockIdx.x * 256 + threadIdx.x;    // over B*N*D
  int d = idx & (DDD - 1);
  int n = (idx >> 8) & (NNN - 1);
  int b = idx >> 19;
  int h = d >> 5;
  int hd = d & 31;
  vt[((size_t)(b * HHH + h) * HDD + hd) * (size_t)NNN + n] = v[idx];
}

// ---------------- LayerNorm: f32 row(256) -> bf16 --------------------------
__global__ __launch_bounds__(256)
void ln_kernel(const float* __restrict__ x, const float* __restrict__ g,
               const float* __restrict__ be, bf16_t* __restrict__ out) {
  __shared__ float red[256];
  int row = blockIdx.x;
  int t   = threadIdx.x;
  float v = x[(size_t)row * DDD + t];
  red[t] = v; __syncthreads();
#pragma unroll
  for (int s = 128; s > 0; s >>= 1) { if (t < s) red[t] += red[t + s]; __syncthreads(); }
  float mu = red[0] * (1.0f / DDD);
  __syncthreads();
  float d = v - mu;
  red[t] = d * d; __syncthreads();
#pragma unroll
  for (int s = 128; s > 0; s >>= 1) { if (t < s) red[t] += red[t + s]; __syncthreads(); }
  float var = red[0] * (1.0f / DDD);
  float y = d * rsqrtf(var + EPSLN) * g[t] + be[t];
  out[(size_t)row * DDD + t] = (bf16_t)y;
}

// ---------------- generic WMMA GEMM:  C = epi(A * W^T + bias) --------------
// A: MxK bf16 row-major.  WT: NxK bf16 (W transposed).
// Block = 256 threads = 8 waves; tile 32(M) x 256(N); each wave 16x64.
// EPI 0: store bf16.  EPI 1: exact GELU -> bf16.  EPI 2: f32 out = res + acc.
template <int EPI>
__global__ __launch_bounds__(256)
void gemm_kernel(const bf16_t* __restrict__ A, const bf16_t* __restrict__ WT,
                 const float* __restrict__ bias, const float* __restrict__ res,
                 float* __restrict__ outf, bf16_t* __restrict__ outb,
                 int M, int Nout, int K) {
  int wave = threadIdx.x >> 5;
  int lane = threadIdx.x & 31;
  int wm = wave & 1, wn = wave >> 1;
  int mtile = blockIdx.x * 32 + wm * 16;
  int nbase = blockIdx.y * 256 + wn * 64;

  v8f acc[4] = {v8f{}, v8f{}, v8f{}, v8f{}};
  for (int k0 = 0; k0 < K; k0 += 32) {
    const bf16_t* ap = A + (size_t)mtile * K + k0;
    if (k0 + 32 < K) __builtin_prefetch(ap + 32, 0, 3);   // global_prefetch_b8
    v16bf af = load_frag_g(ap, K);
#pragma unroll
    for (int j = 0; j < 4; ++j) {
      v16bf bf = load_frag_g(WT + (size_t)(nbase + j * 16) * K + k0, K);
      acc[j] = wmma_bf16(af, bf, acc[j]);
    }
  }

  int chalf = (lane >> 4) << 3;   // C layout: vgpr i -> row i (+8 for hi half)
  int ccol  = lane & 15;
#pragma unroll
  for (int j = 0; j < 4; ++j) {
    int col = nbase + j * 16 + ccol;
    float bv = bias[col];
#pragma unroll
    for (int i = 0; i < 8; ++i) {
      int row = mtile + i + chalf;
      float v = acc[j][i] + bv;
      if (EPI == 0) {
        outb[(size_t)row * Nout + col] = (bf16_t)v;
      } else if (EPI == 1) {
        float gg = 0.5f * v * (1.0f + erff(v * 0.70710678118f)); // exact GELU
        outb[(size_t)row * Nout + col] = (bf16_t)gg;
      } else {
        outf[(size_t)row * Nout + col] = res[(size_t)row * Nout + col] + v;
      }
    }
  }
}

// ---------------- masked flash attention -----------------------------------
// One wave owns 16 query rows of one (b,h); streams keys in tiles of 32.
// HD=32 => one WMMA per 16x16 score tile.  Streaming softmax in base-2
// (v_exp_f32 is natively 2^x), fully branchless: masked scores are -inf and
// the running max starts at a finite floor (-1e4, unreachable by real
// q.k/sqrt(HD) scores), so p = exp2(-inf - m) = 0 and corr = exp2(m - m') are
// correct even for fully-masked rows (l stays 0 -> output 0 == nan_to_num).
__global__ __launch_bounds__(256)
void attn_kernel(const bf16_t* __restrict__ q, const bf16_t* __restrict__ k,
                 const bf16_t* __restrict__ vT, const unsigned int* __restrict__ bits,
                 bf16_t* __restrict__ out) {
  __shared__ unsigned short psc[8][16 * 32];
  int wave = threadIdx.x >> 5;
  int lane = threadIdx.x & 31;
  int qblk = blockIdx.x % (NNN / 128);
  int bh   = blockIdx.x / (NNN / 128);
  int b = bh / HHH, h = bh % HHH;
  int qbase = qblk * 128 + wave * 16;
  // 1/sqrt(32) * log2(e): softmax carried out in base-2 domain.
  const float scale2 = 0.17677669529663687f * 1.4426950408889634f;

  v16bf qf = load_frag_g(q + ((size_t)(b * NNN + qbase)) * DDD + h * HDD, DDD);
  v8f o0 = {}; v8f o1 = {};
  float m_i[8], l_i[8];
#pragma unroll
  for (int i = 0; i < 8; ++i) { m_i[i] = -1.0e4f; l_i[i] = 0.f; }

  int chalf = (lane >> 4) << 3;
  int ccol  = lane & 15;
  const unsigned int* bitsb = bits + (size_t)b * NNN * (NNN / 32);
  const bf16_t*       vbase = vT + (size_t)(b * HHH + h) * HDD * NNN;

  for (int kb = 0; kb < NNN; kb += 32) {
    v16bf kf0 = load_frag_g(k + ((size_t)(b * NNN + kb)) * DDD + h * HDD, DDD);
    v16bf kf1 = load_frag_g(k + ((size_t)(b * NNN + kb + 16)) * DDD + h * HDD, DDD);
    v8f s0 = wmma_bf16(qf, kf0, v8f{});
    v8f s1 = wmma_bf16(qf, kf1, v8f{});
    int wrd = kb >> 5;                         // one u32 word == this key tile

#pragma unroll
    for (int i = 0; i < 8; ++i) {
      int qr = qbase + i + chalf;
      unsigned int wm = bitsb[(size_t)qr * (NNN / 32) + wrd];  // bcast in half
      float a0 = ((wm >> ccol) & 1u)        ? s0[i] * scale2 : -INFINITY;
      float a1 = ((wm >> (ccol + 16)) & 1u) ? s1[i] * scale2 : -INFINITY;

      float tmax = row_max16(fmaxf(a0, a1));
      float mnew = fmaxf(m_i[i], tmax);        // >= -1e4, always finite

      float corr = exp2f(m_i[i] - mnew);       // (0,1]
      float p0   = exp2f(a0 - mnew);           // -inf -> 0 for masked
      float p1   = exp2f(a1 - mnew);

      float rsum = row_sum16(p0 + p1);
      l_i[i] = l_i[i] * corr + rsum;
      m_i[i] = mnew;
      o0[i] *= corr; o1[i] *= corr;

      psc[wave][(i + chalf) * 32 + ccol]      = f2bf_bits(p0);
      psc[wave][(i + chalf) * 32 + ccol + 16] = f2bf_bits(p1);
    }
    // Scratch is wave-private: wave-local DS ordering is sufficient.
    __builtin_amdgcn_wave_barrier();
    asm volatile("s_wait_dscnt 0" ::: "memory");
    v16bf pf = load_frag_lds(&psc[wave][0]);
    asm volatile("" ::: "memory");             // keep reads before next writes
    __builtin_amdgcn_wave_barrier();

    v16bf vf0 = load_frag_g(vbase + kb, NNN);                    // d = 0..15
    v16bf vf1 = load_frag_g(vbase + (size_t)16 * NNN + kb, NNN); // d = 16..31
    o0 = wmma_bf16(pf, vf0, o0);
    o1 = wmma_bf16(pf, vf1, o1);
  }

#pragma unroll
  for (int i = 0; i < 8; ++i) {
    int row = qbase + i + chalf;
    float inv = (l_i[i] > 0.f) ? 1.f / l_i[i] : 0.f;  // isolated rows -> 0
    size_t off = ((size_t)(b * NNN + row)) * DDD + h * HDD;
    out[off + ccol]      = (bf16_t)(o0[i] * inv);
    out[off + ccol + 16] = (bf16_t)(o1[i] * inv);
  }
}

// ---------------------------------------------------------------------------
extern "C" void kernel_launch(void* const* d_in, const int* in_sizes, int n_in,
                              void* d_out, int out_size, void* d_ws, size_t ws_size,
                              hipStream_t stream) {
  const float* x   = (const float*)d_in[0];
  const int*   adj = (const int*)d_in[1];
  const float* wq  = (const float*)d_in[2];
  const float* bq  = (const float*)d_in[3];
  const float* wk  = (const float*)d_in[4];
  const float* bk  = (const float*)d_in[5];
  const float* wv  = (const float*)d_in[6];
  const float* bv  = (const float*)d_in[7];
  const float* wo  = (const float*)d_in[8];
  const float* bo  = (const float*)d_in[9];
  const float* g1  = (const float*)d_in[10];
  const float* be1 = (const float*)d_in[11];
  const float* g2  = (const float*)d_in[12];
  const float* be2 = (const float*)d_in[13];
  const float* w1  = (const float*)d_in[14];
  const float* b1  = (const float*)d_in[15];
  const float* w2  = (const float*)d_in[16];
  const float* b2  = (const float*)d_in[17];
  float* out = (float*)d_out;

  const int M = BB * NNN;                      // 8192 rows
  char* ws = (char*)d_ws;                      // needs ~36 MB (fits L2 easily)

  const size_t MB = 1ull << 20;
  bf16_t* xn   = (bf16_t*)(ws + 0);            // 4MB  (reused as h after attn)
  bf16_t* wqT  = (bf16_t*)(ws + 4 * MB);
  bf16_t* wkT  = (bf16_t*)(ws + 4 * MB + (128ull << 10));
  bf16_t* wvT  = (bf16_t*)(ws + 4 * MB + (256ull << 10));
  bf16_t* woT  = (bf16_t*)(ws + 4 * MB + (384ull << 10));
  bf16_t* w1T  = (bf16_t*)(ws + 4 * MB + (512ull << 10));   // 512K
  bf16_t* w2T  = (bf16_t*)(ws + 5 * MB);                    // 512K
  bf16_t* qb   = (bf16_t*)(ws + 6 * MB);       // 4MB
  bf16_t* kb_  = (bf16_t*)(ws + 10 * MB);      // 4MB
  bf16_t* vb   = (bf16_t*)(ws + 14 * MB);      // 4MB
  bf16_t* vTb  = (bf16_t*)(ws + 18 * MB);      // 4MB
  bf16_t* f1   = (bf16_t*)(ws + 6 * MB);       // 16MB, reuses q/k/v/vT region
  bf16_t* aout = (bf16_t*)(ws + 22 * MB);      // 4MB
  float*  x1   = (float*)(ws + 26 * MB);       // 8MB
  unsigned int* abits = (unsigned int*)(ws + 34 * MB); // 2MB packed adjacency

  dim3 blk(256);

  // 1) weights -> bf16 transposed; adjacency -> bitmask
  wtrans_kernel<<<dim3((DDD * DDD + 255) / 256), blk, 0, stream>>>(wq, wqT, DDD, DDD);
  wtrans_kernel<<<dim3((DDD * DDD + 255) / 256), blk, 0, stream>>>(wk, wkT, DDD, DDD);
  wtrans_kernel<<<dim3((DDD * DDD + 255) / 256), blk, 0, stream>>>(wv, wvT, DDD, DDD);
  wtrans_kernel<<<dim3((DDD * DDD + 255) / 256), blk, 0, stream>>>(wo, woT, DDD, DDD);
  wtrans_kernel<<<dim3((DDD * DFF + 255) / 256), blk, 0, stream>>>(w1, w1T, DDD, DFF);
  wtrans_kernel<<<dim3((DFF * DDD + 255) / 256), blk, 0, stream>>>(w2, w2T, DFF, DDD);
  adjpack_kernel<<<dim3((BB * NNN * (NNN / 32)) / 256), blk, 0, stream>>>(adj, abits);

  // 2) LN1
  ln_kernel<<<dim3(M), blk, 0, stream>>>(x, g1, be1, xn);

  // 3) Q/K/V projections (bf16 out, bias fused)
  gemm_kernel<0><<<dim3(M / 32, DDD / 256), blk, 0, stream>>>(xn, wqT, bq, nullptr, nullptr, qb,  M, DDD, DDD);
  gemm_kernel<0><<<dim3(M / 32, DDD / 256), blk, 0, stream>>>(xn, wkT, bk, nullptr, nullptr, kb_, M, DDD, DDD);
  gemm_kernel<0><<<dim3(M / 32, DDD / 256), blk, 0, stream>>>(xn, wvT, bv, nullptr, nullptr, vb,  M, DDD, DDD);

  // 4) V -> per-head transposed layout for the P*V WMMA B operand
  vtrans_kernel<<<dim3((M * DDD) / 256), blk, 0, stream>>>(vb, vTb);

  // 5) masked flash attention
  attn_kernel<<<dim3(BB * HHH * (NNN / 128)), blk, 0, stream>>>(qb, kb_, vTb, abits, aout);

  // 6) output projection + residual -> x1 (f32)
  gemm_kernel<2><<<dim3(M / 32, 1), blk, 0, stream>>>(aout, woT, bo, x, x1, nullptr, M, DDD, DDD);

  // 7) LN2 (reuse xn buffer)
  ln_kernel<<<dim3(M), blk, 0, stream>>>(x1, g2, be2, xn);

  // 8) FFN up + exact GELU
  gemm_kernel<1><<<dim3(M / 32, DFF / 256), blk, 0, stream>>>(xn, w1T, b1, nullptr, nullptr, f1, M, DFF, DDD);

  // 9) FFN down + residual -> final output
  gemm_kernel<2><<<dim3(M / 32, 1), blk, 0, stream>>>(f1, w2T, b2, x1, out, nullptr, M, DDD, DFF);
}
